// MultiScaleSpikingFusion_27350351741615
// MI455X (gfx1250) — matrix operand order repaired and checked by reading, not compiled, not run on hardware
//
#include <hip/hip_runtime.h>
#include <hip/hip_bf16.h>
#include <math.h>

// ---------------------------------------------------------------------------
// MultiScaleSpikingFusion on MI455X (gfx1250, wave32, WMMA bf16 + TDM)
//
// All GEMM stages use v_wmma_f32_16x16x32_bf16.  One wave owns one 16x16
// output tile.  VGPR operand layouts follow CDNA5 ISA:
//   A (16x32, bf16): lane 0-15 row M=lane, K chunks [0..7],[16..23];
//                    lane 16-31 row M=lane-16, K chunks [8..15],[24..31]
//   B (32x16, bf16): lane holds col N=lane%16, K chunks as A:
//                    [0..7],[16..23] (lane<16); [8..15],[24..31] (lane>=16)
//   C/D (16x16,f32): lane<16 -> (m=i, n=lane); lane>=16 -> (m=8+i, n=lane-16)
//
// The TIM scan stages its per-timestep input row (32x480 bf16 = 30720 B,
// contiguous) into LDS with the Tensor Data Mover (tensor_load_to_lds +
// s_wait_tensorcnt), one DMA per step issued by wave 0.
// ---------------------------------------------------------------------------

typedef __bf16 bf16;
typedef __attribute__((ext_vector_type(16))) __bf16 v16bf;
typedef __attribute__((ext_vector_type(8)))  __bf16 v8bf;
typedef __attribute__((ext_vector_type(8)))  float  v8f;

#define TDIM 1024
#define BDIM 32
#define CDIM 384
#define CRDIM 96
#define CFDIM 480
#define NROW (TDIM*BDIM)      // 32768 flattened (t,b) rows
#define EPSBN 1e-5f
#define SPK_TH 0.625f          // VTH * TAU

__device__ __forceinline__ v8f wmma_bf16(v16bf a, v16bf b, v8f c) {
  return __builtin_amdgcn_wmma_f32_16x16x32_bf16(false, a, false, b, (short)0, c,
                                                 false, false);
}

// A-fragment: base points at (row0, k=0); row stride in elements.
__device__ __forceinline__ v16bf frag_a(const bf16* base, int rstride, int k0) {
  unsigned lane = threadIdx.x & 31u;
  const bf16* p = base + (size_t)(lane & 15u) * rstride + k0 + ((lane < 16u) ? 0 : 8);
  v8bf c0 = *(const v8bf*)p;
  v8bf c1 = *(const v8bf*)(p + 16);
  v16bf r;
#pragma unroll
  for (int i = 0; i < 8; ++i) { r[i] = c0[i]; r[i + 8] = c1[i]; }
  return r;
}

// B-fragment from row-major W[n][k]: base points at (n0 row, k=0).
__device__ __forceinline__ v16bf frag_b(const bf16* base, int rstride, int k0) {
  unsigned lane = threadIdx.x & 31u;
  const bf16* p = base + (size_t)(lane & 15u) * rstride + k0 + ((lane < 16u) ? 0 : 8);
  v8bf c0 = *(const v8bf*)p;
  v8bf c1 = *(const v8bf*)(p + 16);
  v16bf r;
#pragma unroll
  for (int i = 0; i < 8; ++i) { r[i] = c0[i]; r[i + 8] = c1[i]; }
  return r;
}

// ----------------------- Tensor Data Mover support -------------------------

#if __has_builtin(__builtin_amdgcn_tensor_load_to_lds) && \
    __has_builtin(__builtin_amdgcn_s_wait_tensorcnt)
#define USE_TDM 1
typedef __attribute__((ext_vector_type(4))) unsigned int v4u;
typedef __attribute__((ext_vector_type(8))) int v8i;
typedef __attribute__((ext_vector_type(4))) int v4i;

// Issue a 1-D TDM copy of `bytes` (multiple of 4, < 256 KB) from global
// memory into LDS byte-offset `lds_off`.  Descriptor layout per
// CDNA5 ISA (D# group0/group1; groups 2/3 unused).
// This toolchain's builtin takes 6 args: (g0 u32x4, g1 i32x8, i32x4, i32x4,
// i32x8, cpol).
__device__ __forceinline__ void tdm_load_1d(const void* gsrc, unsigned lds_off,
                                            unsigned bytes) {
  unsigned long long ga = (unsigned long long)(size_t)gsrc;
  unsigned n4 = bytes >> 2;                          // 4-byte elements
  v4u g0 = {};
  g0[0] = 1u;                                        // count=1, user D#
  g0[1] = lds_off;                                   // D#.lds_addr
  g0[2] = (unsigned)ga;                              // global_addr[31:0]
  g0[3] = (unsigned)((ga >> 32) & 0x01ffffffu)       // global_addr[56:32]
        | (2u << 30);                                // type = 2 ("image")
  v8i g1 = {};
  g1[0] = (int)(2u << 16);                           // data_size=4B, mask=0
  g1[1] = (int)((n4 & 0xffffu) << 16);               // tensor_dim0[15:0]
  g1[2] = (int)((n4 >> 16) & 0xffffu);               // tensor_dim0[31:16]
  g1[3] = (int)((n4 & 0xffffu) << 16);               // tile_dim0 (1-D tile)
  g1[5] = (int)n4;                                   // tensor_dim0_stride
  v4i gz = {};
  v8i gz8 = {};
  __builtin_amdgcn_tensor_load_to_lds(g0, g1, gz, gz, gz8, 0);
}
#else
#define USE_TDM 0
#endif

// --------------------------- packing kernels -------------------------------

__global__ void k_cvt_bf16(const float* __restrict__ src, bf16* __restrict__ dst, int n) {
  int i = blockIdx.x * blockDim.x + threadIdx.x;
  if (i < n) dst[i] = (bf16)src[i];
}

// (Cl, C, k) -> per-tap (k, Cl, C) bf16
__global__ void k_pack_tap(const float* __restrict__ src, bf16* __restrict__ dst, int ktaps) {
  int i = blockIdx.x * blockDim.x + threadIdx.x;
  int total = ktaps * 128 * CDIM;
  if (i >= total) return;
  int j = i / (128 * CDIM);
  int rem = i % (128 * CDIM);
  int o = rem / CDIM, c = rem % CDIM;
  dst[i] = (bf16)src[(o * CDIM + c) * ktaps + j];
}

// W_tim[:, :, 1] -> Wm bf16 (480x480)
__global__ void k_pack_wm(const float* __restrict__ wtim, bf16* __restrict__ wm) {
  int i = blockIdx.x * blockDim.x + threadIdx.x;
  if (i < CFDIM * CFDIM) wm[i] = (bf16)wtim[i * 3 + 1];
}

// --------------------------- local spiking convs ---------------------------

struct LocalBN { const float *b, *g, *beta, *rm, *rv; };
struct LocalBN3 { LocalBN c[3]; };

// grid (2048, 24), block 32. nt: 0-7 conv0, 8-15 conv1, 16-23 conv2.
__global__ __launch_bounds__(32) void k_conv(const bf16* __restrict__ xb,
                                             const bf16* __restrict__ wl,
                                             LocalBN3 bnp,
                                             bf16* __restrict__ fused) {
  int mt = blockIdx.x, nt = blockIdx.y;
  int gsel = nt >> 3, oct = nt & 7;
  int t = mt >> 1;                       // 16-row tile lies within one t
  int kk = (gsel == 0) ? 3 : (gsel == 1) ? 5 : 7;
  int dil = gsel + 1;
  int pad = (kk - 1) * dil / 2;
  size_t wbase = (gsel == 0) ? 0 : (gsel == 1) ? (size_t)3 * 128 * CDIM
                                               : (size_t)8 * 128 * CDIM;
  const bf16* wg = wl + wbase;
  v8f acc = {};
  for (int j = 0; j < kk; ++j) {
    int tp = t + j * dil - pad;
    if (tp < 0 || tp >= TDIM) continue;  // uniform per block: EXEC stays full
    const bf16* abase = xb + ((size_t)tp * BDIM + (mt & 1) * 16) * CDIM;
    const bf16* wt = wg + (size_t)j * 128 * CDIM + (size_t)(oct * 16) * CDIM;
#pragma unroll
    for (int ks = 0; ks < CDIM / 32; ++ks)
      acc = wmma_bf16(frag_a(abase, CDIM, ks * 32), frag_b(wt, CDIM, ks * 32), acc);
  }
  unsigned lane = threadIdx.x & 31u;
  int oc = oct * 16 + (lane & 15);
  const LocalBN& P = bnp.c[gsel];
  float sc = P.g[oc] * rsqrtf(P.rv[oc] + EPSBN);
  float sh = (P.b[oc] - P.rm[oc]) * sc + P.beta[oc];
  int col = gsel * 128 + oc;
  size_t rbase = (size_t)mt * 16 * CFDIM;
#pragma unroll
  for (int i = 0; i < 8; ++i) {
    int m = (lane < 16u) ? i : 8 + i;
    float y = acc[i] * sc + sh;
    fused[rbase + (size_t)m * CFDIM + col] = (bf16)((y >= SPK_TH) ? 1.0f : 0.0f);
  }
}

// --------------------------- reduced = W_cr @ x ----------------------------

// grid (2048, 6), block 32 -> reduced fp32 (row, 96)
__global__ __launch_bounds__(32) void k_red(const bf16* __restrict__ xb,
                                            const bf16* __restrict__ wcr,
                                            const float* __restrict__ bcr,
                                            float* __restrict__ red) {
  int mt = blockIdx.x, ct = blockIdx.y;
  const bf16* abase = xb + (size_t)mt * 16 * CDIM;
  const bf16* wb = wcr + (size_t)(ct * 16) * CDIM;
  v8f acc = {};
#pragma unroll
  for (int ks = 0; ks < CDIM / 32; ++ks)
    acc = wmma_bf16(frag_a(abase, CDIM, ks * 32), frag_b(wb, CDIM, ks * 32), acc);
  unsigned lane = threadIdx.x & 31u;
  int c = ct * 16 + (lane & 15);
  float bias = bcr[c];
#pragma unroll
  for (int i = 0; i < 8; ++i) {
    int m = (lane < 16u) ? i : 8 + i;
    red[(size_t)(mt * 16 + m) * CRDIM + c] = acc[i] + bias;
  }
}

// normalize rows of reduced; emit raw bf16 and normalized bf16
__global__ void k_norm(const float* __restrict__ red, bf16* __restrict__ rb,
                       bf16* __restrict__ nn) {
  int r = blockIdx.x * blockDim.x + threadIdx.x;
  if (r >= NROW) return;
  float ss = 0.f;
  for (int c = 0; c < CRDIM; ++c) {
    float v = red[(size_t)r * CRDIM + c];
    ss += v * v;
    rb[(size_t)r * CRDIM + c] = (bf16)v;
  }
  float inv = 1.0f / fmaxf(sqrtf(ss), 1e-12f);
  for (int c = 0; c < CRDIM; ++c)
    nn[(size_t)r * CRDIM + c] = (bf16)(red[(size_t)r * CRDIM + c] * inv);
}

// gcn = W_gcn @ reduced, written transposed to gcnT[b][c][t]
__global__ __launch_bounds__(32) void k_gcn(const bf16* __restrict__ rb,
                                            const bf16* __restrict__ wgcn,
                                            const float* __restrict__ bgcn,
                                            bf16* __restrict__ gcnT) {
  int mt = blockIdx.x, ct = blockIdx.y;
  const bf16* abase = rb + (size_t)mt * 16 * CRDIM;
  const bf16* wb = wgcn + (size_t)(ct * 16) * CRDIM;
  v8f acc = {};
#pragma unroll
  for (int ks = 0; ks < CRDIM / 32; ++ks)
    acc = wmma_bf16(frag_a(abase, CRDIM, ks * 32), frag_b(wb, CRDIM, ks * 32), acc);
  unsigned lane = threadIdx.x & 31u;
  int c = ct * 16 + (lane & 15);
  float bias = bgcn[c];
#pragma unroll
  for (int i = 0; i < 8; ++i) {
    int m = (lane < 16u) ? i : 8 + i;
    int r = mt * 16 + m;
    int t = r >> 5, b = r & 31;
    gcnT[((size_t)b * CRDIM + c) * TDIM + t] = (bf16)(acc[i] + bias);
  }
}

// -------------- sim + row softmax + positional adjacency -> adjT -----------

// grid (64, 32), block 256 (8 waves). Each block: 16 t-rows, all 1024 s.
__global__ __launch_bounds__(256) void k_simadj(const bf16* __restrict__ nn,
                                                bf16* __restrict__ adjT) {
  __shared__ float sred[8][16];
  __shared__ float rowmax[16];
  __shared__ float rowsum[16];
  int tb = blockIdx.x, bb = blockIdx.y;
  int t0 = tb * 16;
  int w = threadIdx.x >> 5;
  unsigned lane = threadIdx.x & 31u;
  const float E1 = 0.36787944117144233f;      // exp(-1)
  const float INV1ME = 1.5819767068693265f;   // 1/(1-exp(-1))

  const bf16* abase = nn + ((size_t)t0 * BDIM + bb) * CRDIM;   // rstride 32*96
  v16bf afr[3];
#pragma unroll
  for (int ks = 0; ks < 3; ++ks) afr[ks] = frag_a(abase, BDIM * CRDIM, ks * 32);

  v8f dfr[8];
#pragma unroll
  for (int st = 0; st < 8; ++st) {
    int s0 = (w * 8 + st) * 16;
    const bf16* bbase = nn + ((size_t)s0 * BDIM + bb) * CRDIM;
    v8f acc = {};
#pragma unroll
    for (int ks = 0; ks < 3; ++ks)
      acc = wmma_bf16(afr[ks], frag_b(bbase, BDIM * CRDIM, ks * 32), acc);
    dfr[st] = acc;
  }

  int mlo = (lane < 16u) ? 0 : 8;
  // ---- row max ----
  float pmax[8];
#pragma unroll
  for (int i = 0; i < 8; ++i) {
    float m = -1e30f;
#pragma unroll
    for (int st = 0; st < 8; ++st) m = fmaxf(m, dfr[st][i]);
    pmax[i] = m;
  }
#pragma unroll
  for (int off = 1; off < 16; off <<= 1)
#pragma unroll
    for (int i = 0; i < 8; ++i) pmax[i] = fmaxf(pmax[i], __shfl_xor(pmax[i], off, 32));
  if (lane == 0u)  for (int i = 0; i < 8; ++i) sred[w][i] = pmax[i];
  if (lane == 16u) for (int i = 0; i < 8; ++i) sred[w][8 + i] = pmax[i];
  __syncthreads();
  if (threadIdx.x < 16) {
    float m = -1e30f;
    for (int ww = 0; ww < 8; ++ww) m = fmaxf(m, sred[ww][threadIdx.x]);
    rowmax[threadIdx.x] = m;
  }
  __syncthreads();
  // ---- row sum of exp ----
  float psum[8];
#pragma unroll
  for (int i = 0; i < 8; ++i) {
    float rm = rowmax[mlo + i], s = 0.f;
#pragma unroll
    for (int st = 0; st < 8; ++st) s += expf(dfr[st][i] - rm);
    psum[i] = s;
  }
#pragma unroll
  for (int off = 1; off < 16; off <<= 1)
#pragma unroll
    for (int i = 0; i < 8; ++i) psum[i] += __shfl_xor(psum[i], off, 32);
  if (lane == 0u)  for (int i = 0; i < 8; ++i) sred[w][i] = psum[i];
  if (lane == 16u) for (int i = 0; i < 8; ++i) sred[w][8 + i] = psum[i];
  __syncthreads();
  if (threadIdx.x < 16) {
    float s = 0.f;
    for (int ww = 0; ww < 8; ++ww) s += sred[ww][threadIdx.x];
    rowsum[threadIdx.x] = s;
  }
  __syncthreads();

  float rmx[8], rsm[8], dsm[8];
#pragma unroll
  for (int i = 0; i < 8; ++i) {
    int m = mlo + i, tg = t0 + m;
    rmx[i] = rowmax[m];
    rsm[i] = rowsum[m];
    // closed-form row sum of exp(-|t-s|) over s=0..1023
    float a = (1.0f - expf(-(float)(tg + 1))) * INV1ME;
    float b = E1 * (1.0f - expf(-(float)(TDIM - 1 - tg))) * INV1ME;
    dsm[i] = a + b;
  }
#pragma unroll
  for (int st = 0; st < 8; ++st) {
    int sg = (w * 8 + st) * 16 + (lane & 15);
    v8bf ov;
#pragma unroll
    for (int i = 0; i < 8; ++i) {
      int tg = t0 + mlo + i;
      float val = expf(dfr[st][i] - rmx[i]) / rsm[i]
                + expf(-fabsf((float)(tg - sg))) / dsm[i];
      ov[i] = (bf16)val;
    }
    *(v8bf*)(adjT + ((size_t)bb * TDIM + sg) * TDIM + t0 + mlo) = ov;
  }
}

// ---------------- gsf = LIF(gcn @ adj), K = 1024 ---------------------------

// grid (6, 64, 32), block 32
__global__ __launch_bounds__(32) void k_gsf(const bf16* __restrict__ gcnT,
                                            const bf16* __restrict__ adjT,
                                            bf16* __restrict__ fused) {
  int ct = blockIdx.x, st = blockIdx.y, b = blockIdx.z;
  const bf16* abase = gcnT + ((size_t)b * CRDIM + ct * 16) * TDIM;
  const bf16* bbase = adjT + ((size_t)b * TDIM + st * 16) * TDIM;
  unsigned lane = threadIdx.x & 31u;
  v8f acc = {};
  for (int ks = 0; ks < TDIM / 32; ++ks) {
    __builtin_prefetch(bbase + (size_t)(lane & 15u) * TDIM + (ks + 1) * 32, 0, 1);
    acc = wmma_bf16(frag_a(abase, TDIM, ks * 32), frag_b(bbase, TDIM, ks * 32), acc);
  }
  int s = st * 16 + (lane & 15);
  int mlo = (lane < 16u) ? 0 : 8;
  v8bf sp;
#pragma unroll
  for (int i = 0; i < 8; ++i) sp[i] = (bf16)((acc[i] >= SPK_TH) ? 1.0f : 0.0f);
  *(v8bf*)(fused + ((size_t)s * BDIM + b) * CFDIM + 3 * 128 + ct * 16 + mlo) = sp;
}

// ---------------- TIM sequential scan + fused FC/sigmoid -------------------

// 1 block, 960 threads = 30 waves; wave w owns channels [16w, 16w+16).
__global__ __launch_bounds__(960, 1) void k_tim(const bf16* __restrict__ fused,
                                                const bf16* __restrict__ wm,
                                                const float* __restrict__ btim,
                                                const float* __restrict__ gtim,
                                                const float* __restrict__ betatim,
                                                const float* __restrict__ rmtim,
                                                const float* __restrict__ rvtim,
                                                const float* __restrict__ wfc,
                                                const float* __restrict__ bfc,
                                                float* __restrict__ out) {
  __shared__ bf16 prevLDS[BDIM * CFDIM];   // A operand for next step, 30 KB
  __shared__ float scoreAcc[BDIM];
#if USE_TDM
  __shared__ bf16 curLDS[BDIM * CFDIM];    // TDM-staged fused[t] row, 30 KB
#endif
  int tid = threadIdx.x;
  int wave = tid >> 5;
  unsigned lane = tid & 31u;

  for (int i = tid; i < BDIM * CFDIM; i += 960) prevLDS[i] = (bf16)0.0f;
  if (tid < BDIM) scoreAcc[tid] = 0.0f;

  int n = wave * 16 + (lane & 15);        // fixed output channel per lane
  float sc = gtim[n] * rsqrtf(rvtim[n] + EPSBN);
  float sh = (btim[n] - rmtim[n]) * sc + betatim[n];
  float wf = wfc[n];
  float bfv = bfc[0];
  const bf16* wrow = wm + (size_t)n * CFDIM + ((lane < 16u) ? 0 : 8);

  float v[2][8];
#pragma unroll
  for (int mt = 0; mt < 2; ++mt)
#pragma unroll
    for (int i = 0; i < 8; ++i) v[mt][i] = 0.0f;

#if USE_TDM
  if (wave == 0) {                         // stage fused[0] via TDM
    tdm_load_1d(fused, (unsigned)(size_t)&curLDS[0], BDIM * CFDIM * 2);
    __builtin_amdgcn_s_wait_tensorcnt(0);
  }
#endif
  __syncthreads();

  for (int t = 0; t < TDIM; ++t) {
    float nw[2][8];
#pragma unroll
    for (int mt = 0; mt < 2; ++mt) {
      v8f acc = {};
      const bf16* ab = &prevLDS[(size_t)(mt * 16 + (lane & 15)) * CFDIM +
                                ((lane < 16u) ? 0 : 8)];
#pragma unroll
      for (int ks = 0; ks < CFDIM / 32; ++ks) {
        v8bf c0 = *(const v8bf*)(ab + ks * 32);
        v8bf c1 = *(const v8bf*)(ab + ks * 32 + 16);
        v16bf a;
#pragma unroll
        for (int i = 0; i < 8; ++i) { a[i] = c0[i]; a[i + 8] = c1[i]; }
        v8bf b0 = *(const v8bf*)(wrow + ks * 32);  // Wm from L2 each step
        v8bf b1 = *(const v8bf*)(wrow + ks * 32 + 16);
        v16bf bb;
#pragma unroll
        for (int i = 0; i < 8; ++i) { bb[i] = b0[i]; bb[i + 8] = b1[i]; }
        acc = wmma_bf16(a, bb, acc);
      }
#pragma unroll
      for (int i = 0; i < 8; ++i) {
        int m = mt * 16 + ((lane < 16u) ? i : 8 + i);
        float y = acc[i] * sc + sh;                       // BN(conv + b_tim)
        float vv = v[mt][i];
        vv += (y - vv) * 1.6f;                            // v += (y - v)/tau
        float spike = (vv >= 1.0f) ? 1.0f : 0.0f;
        v[mt][i] = (spike > 0.0f) ? 0.0f : vv;            // hard reset
#if USE_TDM
        float cur = (float)curLDS[(size_t)m * CFDIM + n];
#else
        float cur = (float)fused[((size_t)t * BDIM + m) * CFDIM + n];
#endif
        nw[mt][i] = 0.4f * cur + 0.6f * spike;            // TIM blend
      }
    }
    __syncthreads();   // all reads of prevLDS / curLDS for step t done
#pragma unroll
    for (int mt = 0; mt < 2; ++mt)
#pragma unroll
      for (int i = 0; i < 8; ++i) {
        int m = mt * 16 + ((lane < 16u) ? i : 8 + i);
        prevLDS[(size_t)m * CFDIM + n] = (bf16)nw[mt][i];
        float p = nw[mt][i] * wf;                         // fused FC partial
        p += __shfl_xor(p, 1, 32); p += __shfl_xor(p, 2, 32);
        p += __shfl_xor(p, 4, 32); p += __shfl_xor(p, 8, 32);
        if ((lane & 15u) == 0u) atomicAdd(&scoreAcc[m], p);
      }
#if USE_TDM
    if (wave == 0 && t + 1 < TDIM) {       // stage fused[t+1] via TDM
      tdm_load_1d(fused + (size_t)(t + 1) * BDIM * CFDIM,
                  (unsigned)(size_t)&curLDS[0], BDIM * CFDIM * 2);
      __builtin_amdgcn_s_wait_tensorcnt(0);
    }
#endif
    __syncthreads();   // writes (incl. DMA) visible
    if (tid < BDIM) {
      float sv = scoreAcc[tid] + bfv;
      out[(size_t)t * BDIM + tid] = 1.0f / (1.0f + expf(-sv));
      scoreAcc[tid] = 0.0f;
    }
  }
}

// ------------------------------- host side ---------------------------------

extern "C" void kernel_launch(void* const* d_in, const int* in_sizes, int n_in,
                              void* d_out, int out_size, void* d_ws, size_t ws_size,
                              hipStream_t stream) {
  (void)n_in; (void)out_size; (void)ws_size;
  const float *X, *Wcr, *bcr, *Wgcn, *bgcn;
  const float *Wl[3];
  LocalBN3 bn;
  const float *Wtim, *btim, *gtim, *betatim, *rmtim, *rvtim, *Wfc, *bfc;

  if (in_sizes[0] == TDIM * BDIM * CDIM) {
    // dict insertion-order flatten: x first, then params depth-first
    X = (const float*)d_in[0];
    Wcr = (const float*)d_in[1];  bcr = (const float*)d_in[2];
    Wgcn = (const float*)d_in[3]; bgcn = (const float*)d_in[4];
    for (int g = 0; g < 3; ++g) {
      int base = 5 + 6 * g;                 // W, b, g, beta, rm, rv
      Wl[g] = (const float*)d_in[base];
      bn.c[g].b = (const float*)d_in[base + 1];
      bn.c[g].g = (const float*)d_in[base + 2];
      bn.c[g].beta = (const float*)d_in[base + 3];
      bn.c[g].rm = (const float*)d_in[base + 4];
      bn.c[g].rv = (const float*)d_in[base + 5];
    }
    Wtim = (const float*)d_in[23]; btim = (const float*)d_in[24];
    gtim = (const float*)d_in[25]; betatim = (const float*)d_in[26];
    rmtim = (const float*)d_in[27]; rvtim = (const float*)d_in[28];
    Wfc = (const float*)d_in[29]; bfc = (const float*)d_in[30];
  } else {
    // jax tree_flatten (alphabetical) fallback: params before x
    Wcr = (const float*)d_in[0];  Wfc = (const float*)d_in[1];
    Wgcn = (const float*)d_in[2]; Wtim = (const float*)d_in[3];
    bcr = (const float*)d_in[4];  bfc = (const float*)d_in[5];
    bgcn = (const float*)d_in[6]; btim = (const float*)d_in[7];
    betatim = (const float*)d_in[8]; gtim = (const float*)d_in[9];
    for (int g = 0; g < 3; ++g) {
      int base = 10 + 6 * g;                // W, b, beta, g, rm, rv
      Wl[g] = (const float*)d_in[base];
      bn.c[g].b = (const float*)d_in[base + 1];
      bn.c[g].beta = (const float*)d_in[base + 2];
      bn.c[g].g = (const float*)d_in[base + 3];
      bn.c[g].rm = (const float*)d_in[base + 4];
      bn.c[g].rv = (const float*)d_in[base + 5];
    }
    rmtim = (const float*)d_in[28]; rvtim = (const float*)d_in[29];
    X = (const float*)d_in[30];
  }

  // ---- workspace carve (≈158 MB, all 256-B aligned, L2-resident) ----
  size_t off = 0;
  auto carve = [&](size_t bytes) -> void* {
    void* p = (char*)d_ws + off;
    off += (bytes + 255) & ~(size_t)255;
    return p;
  };
  bf16* xb16  = (bf16*)carve((size_t)NROW * CDIM * 2);
  bf16* wl16  = (bf16*)carve((size_t)15 * 128 * CDIM * 2);
  bf16* wcr16 = (bf16*)carve((size_t)CRDIM * CDIM * 2);
  bf16* wgcn16= (bf16*)carve((size_t)CRDIM * CRDIM * 2);
  bf16* wm16  = (bf16*)carve((size_t)CFDIM * CFDIM * 2);
  float* red  = (float*)carve((size_t)NROW * CRDIM * 4);
  bf16* rb16  = (bf16*)carve((size_t)NROW * CRDIM * 2);
  bf16* n16   = (bf16*)carve((size_t)NROW * CRDIM * 2);
  bf16* gcnT  = (bf16*)carve((size_t)BDIM * CRDIM * TDIM * 2);
  bf16* adjT  = (bf16*)carve((size_t)BDIM * TDIM * TDIM * 2);
  bf16* fused = (bf16*)carve((size_t)NROW * CFDIM * 2);

  // ---- pack / convert ----
  int nx = NROW * CDIM;
  k_cvt_bf16<<<(nx + 255) / 256, 256, 0, stream>>>(X, xb16, nx);
  k_pack_tap<<<(3 * 128 * CDIM + 255) / 256, 256, 0, stream>>>(Wl[0], wl16, 3);
  k_pack_tap<<<(5 * 128 * CDIM + 255) / 256, 256, 0, stream>>>(Wl[1], wl16 + (size_t)3 * 128 * CDIM, 5);
  k_pack_tap<<<(7 * 128 * CDIM + 255) / 256, 256, 0, stream>>>(Wl[2], wl16 + (size_t)8 * 128 * CDIM, 7);
  k_cvt_bf16<<<(CRDIM * CDIM + 255) / 256, 256, 0, stream>>>(Wcr, wcr16, CRDIM * CDIM);
  k_cvt_bf16<<<(CRDIM * CRDIM + 255) / 256, 256, 0, stream>>>(Wgcn, wgcn16, CRDIM * CRDIM);
  k_pack_wm<<<(CFDIM * CFDIM + 255) / 256, 256, 0, stream>>>(Wtim, wm16);

  // ---- pipeline ----
  k_conv<<<dim3(NROW / 16, 24), 32, 0, stream>>>(xb16, wl16, bn, fused);
  k_red <<<dim3(NROW / 16, CRDIM / 16), 32, 0, stream>>>(xb16, wcr16, bcr, red);
  k_norm<<<NROW / 256, 256, 0, stream>>>(red, rb16, n16);
  k_gcn <<<dim3(NROW / 16, CRDIM / 16), 32, 0, stream>>>(rb16, wgcn16, bgcn, gcnT);
  k_simadj<<<dim3(TDIM / 16, BDIM), 256, 0, stream>>>(n16, adjT);
  k_gsf <<<dim3(CRDIM / 16, TDIM / 16, BDIM), 32, 0, stream>>>(gcnT, adjT, fused);
  k_tim <<<1, 960, 0, stream>>>(fused, wm16, btim, gtim, betatim, rmtim, rvtim,
                                Wfc, bfc, (float*)d_out);
}
